// KDAPolicyNetwork_88871463289239
// MI455X (gfx1250) — matrix operand order repaired and checked by reading, not compiled, not run on hardware
//
#include <hip/hip_runtime.h>
#include <hip/hip_fp16.h>

// ---------------------------------------------------------------------------
// KDA policy network for MI455X (gfx1250, wave32, WMMA).
// All heavy GEMMs (expert projections + mHC phi products + W_o) run on
// v_wmma_f32_16x16x32_f16 with f32 accumulate. Weights are pre-packed per call
// into a contiguous-k B-fragment layout (f16, zero-padded) so the GEMM inner loop
// is ds_load_b128 (A from LDS) + global b128 vector loads (B) + WMMA only.
// The serial KDA scan double-buffers its per-timestep operands into LDS with
// GLOBAL_LOAD_ASYNC_TO_LDS_B128 (ASYNCcnt / s_wait_asynccnt pipeline).
// Requires ws_size >= ~130 MB.
// ---------------------------------------------------------------------------

typedef __attribute__((ext_vector_type(16))) _Float16 v16h;
typedef __attribute__((ext_vector_type(8)))  _Float16 v8h;
typedef __attribute__((ext_vector_type(8)))  float    v8f;

#define BB     8
#define NN     4
#define TT     1024
#define DDIM   256
#define HH     8
#define EE     4
#define DK     32
#define DV     32
#define DKP    64
#define HEXP   32
#define ND     1024
#define LR     8
#define DALPHA 51
#define DPG    158
#define BT     8192   // B*T tokens

// packed-weight per-matrix strides (f16 elements): Ktiles*Ntiles*512
#define PHI_S  (32 * 2 * 512)   // K=1024, N=24->32
#define PRE_S  (8 * 16 * 512)   // K=256,  N=256
#define PG1_S  (8 * 10 * 512)   // K=256,  N=158->160
#define PG2_S  (5 * 16 * 512)   // K=158->160, N=256
#define AUP_S  (8 * 4 * 512)    // K=256,  N=51->64
#define ADN_S  (2 * 4 * 512)    // K=51->64, N=64
#define WKV_S  (8 * 2 * 512)    // K=256,  N=32 (per head)
#define LA_S   (8 * 1 * 512)    // K=256,  N=8->16

__device__ __forceinline__ float sigf(float x)  { return 1.f / (1.f + __expf(-x)); }
__device__ __forceinline__ float siluf(float x) { return x * sigf(x); }
__device__ __forceinline__ float softplusf(float x) { return (x > 20.f) ? x : log1pf(__expf(x)); }
__device__ __forceinline__ void  atomAdd(float* p, float v) { unsafeAtomicAdd(p, v); }

// ---- CDNA5 async global->LDS copy (CDNA5 ISA) --------
// Per-lane 16-byte copy, tracked by ASYNCcnt. LDS destination address is the
// low 32 bits of the flat shared-pointer (ISA 10.2).
__device__ __forceinline__ void async_b128(unsigned lds, const float* g) {
  asm volatile("global_load_async_to_lds_b128 %0, %1, off"
               :: "v"(lds), "v"(g) : "memory");
}
__device__ __forceinline__ void async_wait0() {
  asm volatile("s_wait_asynccnt 0" ::: "memory");
}

// ---- WMMA fragment builders (contiguous k order, same for A and B) ----
// A: 16(M) x 32(K) f16 from LDS. lanes 0-15: M=lane, K in {0..15};
// lanes 16-31: M=lane-16, K in {16..31}; VGPR g packs 2 K values.
__device__ __forceinline__ v16h frag_a(const _Float16* A, int lda, int m0, int k0) {
  const int lane = threadIdx.x & 31;
  const _Float16* p = A + (m0 + (lane & 15)) * lda + k0 + ((lane & 16) ? 16 : 0);
  v16h a;
#pragma unroll
  for (int g = 0; g < 8; ++g) {
    const int kk = 2 * g;
    a[2 * g]     = p[kk];
    a[2 * g + 1] = p[kk + 1];
  }
  return a;
}

// B from pre-packed global f16: each 32(K)x16(N) tile is 512 contiguous f16 in
// lane-major order; each lane reads its 16 f16 (32 bytes) with one vector load.
__device__ __forceinline__ v16h frag_bp(const _Float16* P, int kt, int nt, int Ktiles) {
  const int lane = threadIdx.x & 31;
  return *(const v16h*)(P + (((size_t)(nt * Ktiles + kt)) << 9) + ((size_t)lane << 4));
}

// 64-token x N GEMM over K, 8 waves of a 256-thread block each own 16x16 tiles.
template <typename Epi>
__device__ __forceinline__ void gemm_p(const _Float16* A, int lda, int Ktiles,
                                       const _Float16* P, int N, Epi&& epi) {
  const int wave = threadIdx.x >> 5;
  const int lane = threadIdx.x & 31;
  const int nT = (N + 15) >> 4;
  for (int tile = wave; tile < 4 * nT; tile += 8) {
    const int mI = tile & 3, nI = tile >> 2;
    v8f c = {};
    for (int kt = 0; kt < Ktiles; ++kt) {
      v16h a = frag_a(A, lda, mI * 16, kt * 32);
      v16h b = frag_bp(P, kt, nI, Ktiles);
      c = __builtin_amdgcn_wmma_f32_16x16x32_f16(false, a, false, b,
                                                 (short)0, c, false, false);
    }
    const int r0  = mI * 16 + ((lane & 16) ? 8 : 0);
    const int col = nI * 16 + (lane & 15);
    if (col < N) {
#pragma unroll
      for (int vv = 0; vv < 8; ++vv) epi(r0 + vv, col, c[vv]);
    }
  }
}

// ---------------------------------------------------------------------------
// Pack a (K x N) f32 weight view (strides sK,sN) into WMMA B-tile layout, f16,
// zero-padded to Kt*32 x Nt*16.  blockIdx.y = matrix index.
__global__ void k_pack(const float* W, int sK, int sN, int K, int N,
                       int Kt, int Nt, long srcStride, _Float16* dst) {
  const int mat = blockIdx.y;
  const float* Ws = W + (size_t)mat * srcStride;
  _Float16* Ds = dst + (size_t)mat * ((size_t)Kt * Nt * 512);
  const int total = Kt * Nt * 512;
  for (int i = blockIdx.x * blockDim.x + threadIdx.x; i < total;
       i += gridDim.x * blockDim.x) {
    const int tile = i >> 9, rem = i & 511;
    const int lane = rem >> 4, el = rem & 15;
    const int kt = tile % Kt, nt = tile / Kt;
    const int k = kt * 32 + ((lane & 16) ? 16 : 0) + el;
    const int n = nt * 16 + (lane & 15);
    const float v = (k < K && n < N) ? Ws[(size_t)k * sK + (size_t)n * sN] : 0.f;
    Ds[i] = (_Float16)v;
  }
}

// Pack phi_pre(4) | phi_post(4) | phi_res(16) rows -> 24 (pad 32) columns,
// K = 1024, per expert idx (blockIdx.y).
__global__ void k_pack_phi(const float* ppre, const float* ppost, const float* pres,
                           _Float16* dst) {
  const int idx = blockIdx.y;
  _Float16* Ds = dst + (size_t)idx * PHI_S;
  for (int i = blockIdx.x * blockDim.x + threadIdx.x; i < PHI_S;
       i += gridDim.x * blockDim.x) {
    const int tile = i >> 9, rem = i & 511;
    const int lane = rem >> 4, el = rem & 15;
    const int kt = tile % 32, nt = tile / 32;
    const int k = kt * 32 + ((lane & 16) ? 16 : 0) + el;   // 0..1023
    const int n = nt * 16 + (lane & 15);                   // 0..31
    float v = 0.f;
    if (n < 4)       v = ppre [((size_t)idx * 4  + n)        * ND + k];
    else if (n < 8)  v = ppost[((size_t)idx * 4  + (n - 4))  * ND + k];
    else if (n < 24) v = pres [((size_t)idx * 16 + (n - 8))  * ND + k];
    Ds[i] = (_Float16)v;
  }
}

// Kernel: zero-init accumulators (harness poisons ws with 0xAA).
__global__ void k_zero(float* p, size_t n) {
  size_t i = (size_t)blockIdx.x * blockDim.x + threadIdx.x;
  const size_t st = (size_t)gridDim.x * blockDim.x;
  for (; i < n; i += st) p[i] = 0.f;
}

// PoPE phase tables (T x dk): cos/sin of phi_q and phi_k.
__global__ void k_tab(const float* freqs, const float* pdelta,
                      float* cq, float* sq, float* ck, float* sk) {
  const int i = blockIdx.x * blockDim.x + threadIdx.x;
  if (i >= TT * DK) return;
  const int t = i >> 5, c = i & 31;
  const float phi = (float)t * freqs[c];
  const float del = 6.28318530718f * sigf(pdelta[c]);
  cq[i] = cosf(phi);       sq[i] = sinf(phi);
  ck[i] = cosf(phi - del); sk[i] = sinf(phi - del);
}

// Per-token prep: route mean, RMS-base of xs (f16), router top-1 (eval mode).
__global__ void __launch_bounds__(256) k_prep(const float* stream,
                                              const float* qrw, const float* kvrw,
                                              _Float16* xs16, int* qsel, int* kvsel) {
  const int u = blockIdx.x, b = u >> 10, t = u & (TT - 1);
  const int tid = threadIdx.x, lane = tid & 31, wave = tid >> 5;
  __shared__ float rin[DDIM];
  __shared__ float red[8];
  float v[NN]; float ss = 0.f;
#pragma unroll
  for (int i = 0; i < NN; ++i) {
    v[i] = stream[((size_t)(b * NN + i) * TT + t) * DDIM + tid];
    ss += v[i] * v[i];
  }
  rin[tid] = 0.25f * (v[0] + v[1] + v[2] + v[3]);
  for (int m = 16; m >= 1; m >>= 1) ss += __shfl_xor(ss, m, 32);
  if (lane == 0) red[wave] = ss;
  __syncthreads();
  float tot = 0.f;
#pragma unroll
  for (int w = 0; w < 8; ++w) tot += red[w];
  const float scale = rsqrtf(tot * (1.f / (float)ND) + 1e-6f);
#pragma unroll
  for (int i = 0; i < NN; ++i)
    xs16[(size_t)u * ND + i * DDIM + tid] = (_Float16)(v[i] * scale);
  for (int p = wave; p < 16; p += 8) {
    const int hh = p & 7;
    const float* Wb = (p < 8 ? qrw : kvrw) + (size_t)hh * EE * DDIM;
    float best = -1e30f; int bi = 0;
    for (int e = 0; e < EE; ++e) {
      float s = 0.f;
      for (int j = lane; j < DDIM; j += 32) s += rin[j] * Wb[e * DDIM + j];
      for (int m = 16; m >= 1; m >>= 1) s += __shfl_xor(s, m, 32);
      if (s > best) { best = s; bi = e; }
    }
    if (lane == 0) (p < 8 ? qsel : kvsel)[(size_t)u * HH + hh] = bi;
  }
}

// ---------------------------------------------------------------------------
// Fused per-expert kernel: mHC (WMMA) + Sinkhorn + h_e/rmsnorm + all WMMA
// projections with fused epilogues.  grid = (BT/64, HE), block = 256.
__global__ void __launch_bounds__(256) k_expert(
    const _Float16* xs16, const int* qsel, const int* kvsel,
    const _Float16* pPhi, const _Float16* pWpre, const _Float16* pWpg1,
    const _Float16* pWpg2, const _Float16* pAup, const _Float16* pAdn,
    const _Float16* pBup, const _Float16* pWq, const _Float16* pWk,
    const _Float16* pWv, const _Float16* pLAq, const _Float16* pLAk,
    const _Float16* pLAv,
    const float* lBq, const float* lBk, const float* lBv,
    const float* bdn, const float* mnorm,
    const float* bpre, const float* bpost, const float* bres,
    const float* apre, const float* apost, const float* ares, const float* nw,
    const float* cq, const float* sq, const float* ck, const float* sk,
    float* q_h, float* k_h, float* v_h, float* a_h, float* b_h,
    float* acc_pre, float* acc_post, float* acc_Hres, float* acc_Hpost) {
  const int idx = blockIdx.y, h = idx >> 2, e = idx & 3;
  const int u0 = blockIdx.x * 64;
  const int tid = threadIdx.x, lane = tid & 31, wave = tid >> 5;

  __shared__ _Float16 sX[64 * 256];                          // 32 KB
  __shared__ __align__(16) unsigned char sGraw[64 * 160 * 2]; // 20 KB (pg1 / mHC)
  __shared__ _Float16 sU[64 * 64];                           //  8 KB
  __shared__ _Float16 sL[64 * 8];                            //  1 KB (LoRA t1)
  __shared__ float sLB[LR * DK];                             //  1 KB (LoRA B)
  __shared__ float sHpre[64 * 4];
  __shared__ float kmf[64], qmf[64];
  __shared__ int sAny;
  _Float16* sG  = (_Float16*)sGraw;
  float*    sMH = (float*)sGraw;   // 64 x 24 mHC dots (aliased, consumed early)

  // ---- masks + tile early-out ----
  if (tid == 0) sAny = 0;
  __syncthreads();
  if (tid < 64) {
    const int u = u0 + tid;
    const float km = (kvsel[(size_t)u * HH + h] == e) ? 1.f : 0.f;
    const float qm = (qsel[(size_t)u * HH + h] == e) ? 1.f : 0.f;
    kmf[tid] = km; qmf[tid] = qm;
    if (km > 0.f || qm > 0.f) sAny = 1;
  }
  __syncthreads();
  if (!sAny) return;

  // ---- mHC via WMMA: [64 x 24] = xn[64 x 1024] @ phi_pack.
  // 8 waves each own one persistent 16x16 tile (4 m-tiles x 2 n-tiles);
  // A (xn = xs * mhc_norm) staged through sX in 4 chunks of K=256.
  {
    const int mI = wave & 3, nI = wave >> 2;
    const _Float16* Pphi = pPhi + (size_t)idx * PHI_S;
    const float* mn = mnorm + (size_t)idx * ND;
    v8f c = {};
    for (int ch = 0; ch < 4; ++ch) {
      for (int w = tid * 8; w < 64 * 256; w += 256 * 8) {
        const int rr = w >> 8, j = w & 255;
        const v8h xv = *(const v8h*)(xs16 + (size_t)(u0 + rr) * ND + ch * 256 + j);
        v8h o;
#pragma unroll
        for (int i = 0; i < 8; ++i)
          o[i] = (_Float16)((float)xv[i] * mn[ch * 256 + j + i]);
        *(v8h*)(sX + w) = o;
      }
      __syncthreads();
#pragma unroll
      for (int kt = 0; kt < 8; ++kt) {
        v16h a = frag_a(sX, 256, mI * 16, kt * 32);
        v16h b = frag_bp(Pphi, ch * 8 + kt, nI, 32);
        c = __builtin_amdgcn_wmma_f32_16x16x32_f16(false, a, false, b,
                                                   (short)0, c, false, false);
      }
      __syncthreads();
    }
    const int r0  = mI * 16 + ((lane & 16) ? 8 : 0);
    const int col = nI * 16 + (lane & 15);
    if (col < 24) {
#pragma unroll
      for (int vv = 0; vv < 8; ++vv) sMH[(r0 + vv) * 24 + col] = c[vv];
    }
  }
  __syncthreads();

  // ---- H_pre / H_post / Sinkhorn(H_res) per token ----
  if (tid < 64) {
    const int u = u0 + tid;
    const float ap = apre[idx], aq = apost[idx], ar = ares[idx];
#pragma unroll
    for (int i = 0; i < 4; ++i)
      sHpre[tid * 4 + i] = sigf(ap * sMH[tid * 24 + i] + bpre[(size_t)idx * 4 + i]);
    if (qmf[tid] > 0.f) {
#pragma unroll
      for (int i = 0; i < 4; ++i)
        atomAdd(acc_Hpost + (size_t)u * 4 + i,
                2.f * sigf(aq * sMH[tid * 24 + 4 + i] + bpost[(size_t)idx * 4 + i]));
    }
    float M[16];
#pragma unroll
    for (int kq = 0; kq < 16; ++kq)
      M[kq] = __expf(ar * sMH[tid * 24 + 8 + kq] + bres[(size_t)idx * 16 + kq]);
#pragma unroll
    for (int it = 0; it < 6; ++it) {
#pragma unroll
      for (int i = 0; i < 4; ++i) {
        const float rs = 1.f / (M[i*4] + M[i*4+1] + M[i*4+2] + M[i*4+3]);
#pragma unroll
        for (int j = 0; j < 4; ++j) M[i * 4 + j] *= rs;
      }
#pragma unroll
      for (int j = 0; j < 4; ++j) {
        const float cs = 1.f / (M[j] + M[4+j] + M[8+j] + M[12+j]);
#pragma unroll
        for (int i = 0; i < 4; ++i) M[i * 4 + j] *= cs;
      }
    }
    if (kmf[tid] > 0.f) {
#pragma unroll
      for (int kq = 0; kq < 16; ++kq) atomAdd(acc_Hres + (size_t)u * 16 + kq, M[kq]);
    }
  }
  __syncthreads();

  // ---- h_e + rmsnorm -> x tile (f16 in LDS). 4 threads/token, v8h loads. ----
  {
    const int tok = tid >> 2, qq = tid & 3;
    const _Float16* xp = xs16 + (size_t)(u0 + tok) * ND;
    const float h0 = sHpre[tok*4], h1 = sHpre[tok*4+1], h2 = sHpre[tok*4+2], h3 = sHpre[tok*4+3];
    float ssl = 0.f;
    for (int jj = 0; jj < 64; jj += 8) {
      const int j = qq * 64 + jj;
      const v8h x0 = *(const v8h*)(xp + j);
      const v8h x1 = *(const v8h*)(xp + 256 + j);
      const v8h x2 = *(const v8h*)(xp + 512 + j);
      const v8h x3 = *(const v8h*)(xp + 768 + j);
      v8h o;
#pragma unroll
      for (int w2 = 0; w2 < 8; ++w2) {
        const float he = h0 * (float)x0[w2] + h1 * (float)x1[w2]
                       + h2 * (float)x2[w2] + h3 * (float)x3[w2];
        o[w2] = (_Float16)he;
        ssl += he * he;
      }
      *(v8h*)(sX + tok * 256 + j) = o;
    }
    ssl += __shfl_xor(ssl, 1, 32);
    ssl += __shfl_xor(ssl, 2, 32);
    const float scl = rsqrtf(ssl * (1.f / (float)DDIM) + 1e-6f);
    const float* nwp = nw + (size_t)idx * DDIM;
    for (int jj = 0; jj < 64; ++jj) {
      const int j = qq * 64 + jj;
      sX[tok * 256 + j] = (_Float16)((float)sX[tok * 256 + j] * scl * nwp[j]);
    }
  }
  __syncthreads();

  // ---- zero pad buffers (sG aliases sMH: only after mHC consumed) ----
  for (int w = tid; w < 64 * 160; w += 256) sG[w] = (_Float16)0.f;
  for (int w = tid; w < 64 * 64; w += 256)  sU[w] = (_Float16)0.f;
  __syncthreads();

  const size_t BH = (size_t)h * BT;

  // pg1: x @ W_pg1 -> silu -> sG
  gemm_p(sX, 256, 8, pWpg1 + (size_t)idx * PG1_S, DPG,
         [&](int rr, int cc, float vl) { sG[rr * 160 + cc] = (_Float16)siluf(vl); });
  __syncthreads();
  // pg2: silu(pg1) @ W_pg2 -> acc_post (q-masked)
  gemm_p(sG, 160, 5, pWpg2 + (size_t)idx * PG2_S, DDIM,
         [&](int rr, int cc, float vl) {
           if (qmf[rr] > 0.f) atomAdd(acc_post + (size_t)(u0 + rr) * DDIM + cc, vl);
         });
  // W_pre: x @ W_pre -> acc_pre (kv-masked)
  gemm_p(sX, 256, 8, pWpre + (size_t)idx * PRE_S, HH * DV,
         [&](int rr, int cc, float vl) {
           if (kmf[rr] > 0.f) atomAdd(acc_pre + (size_t)(u0 + rr) * (HH * DV) + cc, vl);
         });
  // alpha_up -> silu -> sU
  gemm_p(sX, 256, 8, pAup + (size_t)idx * AUP_S, DALPHA,
         [&](int rr, int cc, float vl) { sU[rr * 64 + cc] = (_Float16)siluf(vl); });
  __syncthreads();
  // alpha_down: sigmoid -> a_h (kv-masked plain store; masks disjoint over e)
  gemm_p(sU, 64, 2, pAdn + (size_t)idx * ADN_S, DKP,
         [&](int rr, int cc, float vl) {
           if (kmf[rr] > 0.f) a_h[(BH + u0 + rr) * DKP + cc] = sigf(vl);
         });
  __syncthreads();
  // beta_up -> silu -> sU (reuse)
  gemm_p(sX, 256, 8, pBup + (size_t)idx * AUP_S, DALPHA,
         [&](int rr, int cc, float vl) { sU[rr * 64 + cc] = (_Float16)siluf(vl); });
  __syncthreads();
  if (tid < 64 && kmf[tid] > 0.f) {
    const float* bd = bdn + (size_t)idx * DALPHA;
    float s = 0.f;
    for (int kq = 0; kq < DALPHA; ++kq) s += (float)sU[tid * 64 + kq] * bd[kq];
    b_h[BH + u0 + tid] = sigf(s);
  }
  // ---- K path: LoRA rank-8 stage-1, then Wk + fused rank-8 add + PoPE ----
  gemm_p(sX, 256, 8, pLAk + (size_t)idx * LA_S, LR,
         [&](int rr, int cc, float vl) { sL[rr * 8 + cc] = (_Float16)vl; });
  for (int w = tid; w < LR * DK; w += 256) sLB[w] = lBk[(size_t)idx * LR * DK + w];
  __syncthreads();
  gemm_p(sX, 256, 8, pWk + (size_t)h * WKV_S, DK,
         [&](int rr, int cc, float vl) {
           if (kmf[rr] <= 0.f) return;
           float acc = vl;
#pragma unroll
           for (int r2 = 0; r2 < LR; ++r2) acc += (float)sL[rr * 8 + r2] * sLB[r2 * DK + cc];
           const float mu = softplusf(acc);
           const int u = u0 + rr, t = u & (TT - 1);
           const size_t base = (BH + u) * DKP;
           k_h[base + cc]      = mu * ck[t * DK + cc];
           k_h[base + DK + cc] = mu * sk[t * DK + cc];
         });
  __syncthreads();
  // ---- V path ----
  gemm_p(sX, 256, 8, pLAv + (size_t)idx * LA_S, LR,
         [&](int rr, int cc, float vl) { sL[rr * 8 + cc] = (_Float16)vl; });
  for (int w = tid; w < LR * DV; w += 256) sLB[w] = lBv[(size_t)idx * LR * DV + w];
  __syncthreads();
  gemm_p(sX, 256, 8, pWv + (size_t)h * WKV_S, DV,
         [&](int rr, int cc, float vl) {
           if (kmf[rr] <= 0.f) return;
           float acc = vl;
#pragma unroll
           for (int r2 = 0; r2 < LR; ++r2) acc += (float)sL[rr * 8 + r2] * sLB[r2 * DV + cc];
           v_h[(BH + u0 + rr) * DV + cc] = acc;
         });
  // ---- Q path (expert 0 is the no-op expert for Q) ----
  if (e > 0) {
    __syncthreads();
    gemm_p(sX, 256, 8, pLAq + (size_t)idx * LA_S, LR,
           [&](int rr, int cc, float vl) { sL[rr * 8 + cc] = (_Float16)vl; });
    for (int w = tid; w < LR * DK; w += 256) sLB[w] = lBq[(size_t)idx * LR * DK + w];
    __syncthreads();
    gemm_p(sX, 256, 8, pWq + (size_t)h * WKV_S, DK,
           [&](int rr, int cc, float vl) {
             if (qmf[rr] <= 0.f) return;
             float acc = vl;
#pragma unroll
             for (int r2 = 0; r2 < LR; ++r2) acc += (float)sL[rr * 8 + r2] * sLB[r2 * DK + cc];
             const float mu = softplusf(acc);
             const int u = u0 + rr, t = u & (TT - 1);
             const size_t base = (BH + u) * DKP;
             q_h[base + cc]      = mu * cq[t * DK + cc];
             q_h[base + DK + cc] = mu * sq[t * DK + cc];
           });
  }
}

// ---------------------------------------------------------------------------
// KDA rank-1 recurrent scan. 64 blocks = (b,h); S[64x32] in registers.
// Per-timestep operands (q/k/a 64, v 32, b 1) are double-buffered into LDS in
// 16-timestep chunks with GLOBAL_LOAD_ASYNC_TO_LDS_B128: chunk c+1 is issued
// before chunk c is processed, so copy latency hides behind recurrence math,
// and the 32 column-threads read each k/a value once from LDS instead of 32x
// from global.  ASYNCcnt discipline: s_wait_asynccnt 0 + barrier per chunk.
#define SCH 16   // timesteps per chunk
__global__ void __launch_bounds__(256) k_scan(const float* q_h, const float* k_h,
                                              const float* v_h, const float* a_h,
                                              const float* b_h, float* out_cat) {
  const int bh = blockIdx.x, b = bh >> 3, h = bh & 7;
  const int tid = threadIdx.x, e = tid >> 3, g = tid & 7;
  const size_t BH = (size_t)h * BT;
  __shared__ __align__(16) float sKb[2][SCH][DKP];
  __shared__ __align__(16) float sAb[2][SCH][DKP];
  __shared__ __align__(16) float sQb[2][SCH][DKP];
  __shared__ __align__(16) float sVb[2][SCH][DV];
  __shared__ __align__(16) float sBb[2][SCH];

  auto issue = [&](int buf, int c) {
    const size_t base = BH + ((size_t)b << 10) + (size_t)c * SCH;
    // k/a/q: SCH*64 = 1024 floats each (contiguous) -> 256 lanes x 16B
    async_b128((unsigned)(size_t)&sKb[buf][0][0] + tid * 16, k_h + base * DKP + tid * 4);
    async_b128((unsigned)(size_t)&sAb[buf][0][0] + tid * 16, a_h + base * DKP + tid * 4);
    async_b128((unsigned)(size_t)&sQb[buf][0][0] + tid * 16, q_h + base * DKP + tid * 4);
    if (tid < (SCH * DV) / 4)
      async_b128((unsigned)(size_t)&sVb[buf][0][0] + tid * 16, v_h + base * DV + tid * 4);
    if (tid < SCH / 4)
      async_b128((unsigned)(size_t)&sBb[buf][0] + tid * 16, b_h + base + tid * 4);
  };

  float S[8] = {0.f, 0.f, 0.f, 0.f, 0.f, 0.f, 0.f, 0.f};
  issue(0, 0);
  async_wait0();
  __syncthreads();
  for (int c = 0; c < TT / SCH; ++c) {
    const int buf = c & 1;
    if (c + 1 < TT / SCH) issue(1 - buf, c + 1);
    const int u00 = (b << 10) + c * SCH;
    for (int tt = 0; tt < SCH; ++tt) {
      float kx[8], ax[8], qx[8];
      *(float4*)&kx[0] = *(const float4*)&sKb[buf][tt][g * 8];
      *(float4*)&kx[4] = *(const float4*)&sKb[buf][tt][g * 8 + 4];
      *(float4*)&ax[0] = *(const float4*)&sAb[buf][tt][g * 8];
      *(float4*)&ax[4] = *(const float4*)&sAb[buf][tt][g * 8 + 4];
      *(float4*)&qx[0] = *(const float4*)&sQb[buf][tt][g * 8];
      *(float4*)&qx[4] = *(const float4*)&sQb[buf][tt][g * 8 + 4];
      const float vv = sVb[buf][tt][e];
      const float bb = sBb[buf][tt];
      float part = 0.f;
#pragma unroll
      for (int i = 0; i < 8; ++i) { S[i] *= ax[i]; part += kx[i] * S[i]; }
      part += __shfl_xor(part, 1, 32);
      part += __shfl_xor(part, 2, 32);
      part += __shfl_xor(part, 4, 32);
      const float coef = bb * (vv - part);
      float po = 0.f;
#pragma unroll
      for (int i = 0; i < 8; ++i) { S[i] += coef * kx[i]; po += qx[i] * S[i]; }
      po += __shfl_xor(po, 1, 32);
      po += __shfl_xor(po, 2, 32);
      po += __shfl_xor(po, 4, 32);
      if (g == 0) out_cat[(size_t)(u00 + tt) * (HH * DV) + h * DV + e] = po;
    }
    async_wait0();   // chunk c+1 copies complete (issued a full chunk ago)
    __syncthreads();
  }
}

// y = (sigmoid(acc_pre) * out_cat) @ W_o^T with post-gate. WMMA on packed W_o.
__global__ void __launch_bounds__(256) k_y(const float* acc_pre, const float* out_cat,
                                           const float* acc_post, const _Float16* pWo,
                                           float* y) {
  __shared__ _Float16 sX[64 * 256];
  const int u0 = blockIdx.x * 64, tid = threadIdx.x;
  for (int w = tid; w < 64 * 256; w += 256) {
    const int rr = w >> 8, j = w & 255;
    const size_t p = (size_t)(u0 + rr) * 256 + j;
    sX[w] = (_Float16)(out_cat[p] * sigf(acc_pre[p]));
  }
  __syncthreads();
  gemm_p(sX, 256, 8, pWo, 256,
         [&](int rr, int cc, float vl) {
           const size_t p = (size_t)(u0 + rr) * 256 + cc;
           y[p] = vl * sigf(acc_post[p]);
         });
}

// out[b,i,t,:] = sum_j Hres[i,j]/H * stream[b,j,t,:] + Hpost[i]/H * y
__global__ void __launch_bounds__(256) k_out(const float* stream, const float* acc_Hres,
                                             const float* acc_Hpost, const float* y,
                                             float* out) {
  const int u = blockIdx.x, dd = threadIdx.x;
  const int b = u >> 10, t = u & (TT - 1);
  const float inv = 1.f / (float)HH;
  const float yv = y[(size_t)u * DDIM + dd];
  float sv[NN];
#pragma unroll
  for (int j = 0; j < NN; ++j)
    sv[j] = stream[((size_t)(b * NN + j) * TT + t) * DDIM + dd];
#pragma unroll
  for (int i = 0; i < NN; ++i) {
    float acc = acc_Hpost[(size_t)u * 4 + i] * inv * yv;
#pragma unroll
    for (int j = 0; j < NN; ++j)
      acc += acc_Hres[(size_t)u * 16 + i * 4 + j] * inv * sv[j];
    out[((size_t)(b * NN + i) * TT + t) * DDIM + dd] = acc;
  }
}

// ---------------------------------------------------------------------------
extern "C" void kernel_launch(void* const* d_in, const int* in_sizes, int n_in,
                              void* d_out, int out_size, void* d_ws, size_t ws_size,
                              hipStream_t stream) {
  (void)in_sizes; (void)n_in; (void)out_size; (void)ws_size;
  const float* p_stream = (const float*)d_in[0];
  const float* Wq    = (const float*)d_in[1];
  const float* Wk    = (const float*)d_in[2];
  const float* Wv    = (const float*)d_in[3];
  const float* qrw   = (const float*)d_in[4];
  const float* kvrw  = (const float*)d_in[5];
  const float* freqs = (const float*)d_in[6];
  const float* pdel  = (const float*)d_in[7];
  const float* lAq   = (const float*)d_in[8];
  const float* lBq   = (const float*)d_in[9];
  const float* lAk   = (const float*)d_in[10];
  const float* lBk   = (const float*)d_in[11];
  const float* lAv   = (const float*)d_in[12];
  const float* lBv   = (const float*)d_in[13];
  const float* aup   = (const float*)d_in[14];
  const float* adn   = (const float*)d_in[15];
  const float* bup   = (const float*)d_in[16];
  const float* bdn   = (const float*)d_in[17];
  const float* mnorm = (const float*)d_in[18];
  const float* ppre  = (const float*)d_in[19];
  const float* ppost = (const float*)d_in[20];
  const float* pres  = (const float*)d_in[21];
  const float* bpre  = (const float*)d_in[22];
  const float* bpost = (const float*)d_in[23];
  const float* bres  = (const float*)d_in[24];
  const float* apre  = (const float*)d_in[25];
  const float* apost = (const float*)d_in[26];
  const float* ares  = (const float*)d_in[27];
  const float* nw    = (const float*)d_in[28];
  const float* Wpre  = (const float*)d_in[29];
  const float* Wpg1  = (const float*)d_in[30];
  const float* Wpg2  = (const float*)d_in[31];
  const float* W_o   = (const float*)d_in[32];
  float* out = (float*)d_out;

  char* ws = (char*)d_ws;
  size_t off = 0;
  auto alloc = [&](size_t bytes) -> char* {
    const size_t a = (off + 255) & ~(size_t)255;
    off = a + bytes;
    return ws + a;
  };
  _Float16* xs16 = (_Float16*)alloc((size_t)BT * ND * 2);
  int* qsel  = (int*)alloc((size_t)BT * HH * 4);
  int* kvsel = (int*)alloc((size_t)BT * HH * 4);
  float* cq = (float*)alloc((size_t)TT * DK * 4);
  float* sq = (float*)alloc((size_t)TT * DK * 4);
  float* ck = (float*)alloc((size_t)TT * DK * 4);
  float* sk = (float*)alloc((size_t)TT * DK * 4);
  // zero-init region (contiguous; every size is a multiple of 256 bytes)
  float* q_h      = (float*)alloc((size_t)HH * BT * DKP * 4);
  float* acc_pre  = (float*)alloc((size_t)BT * 256 * 4);
  float* acc_post = (float*)alloc((size_t)BT * 256 * 4);
  float* acc_Hres = (float*)alloc((size_t)BT * 16 * 4);
  float* acc_Hpost= (float*)alloc((size_t)BT * 4 * 4);
  const size_t zcount = ((char*)acc_Hpost + (size_t)BT * 4 * 4 - (char*)q_h) / 4;
  float* k_h = (float*)alloc((size_t)HH * BT * DKP * 4);
  float* a_h = (float*)alloc((size_t)HH * BT * DKP * 4);
  float* v_h = (float*)alloc((size_t)HH * BT * DV * 4);
  float* b_h = (float*)alloc((size_t)HH * BT * 4);
  float* out_cat = (float*)alloc((size_t)BT * 256 * 4);
  float* yb      = (float*)alloc((size_t)BT * 256 * 4);
  // packed f16 weights (B-fragment layout)
  _Float16* pPhi  = (_Float16*)alloc((size_t)HEXP * PHI_S * 2);
  _Float16* pWpre = (_Float16*)alloc((size_t)HEXP * PRE_S * 2);
  _Float16* pWpg1 = (_Float16*)alloc((size_t)HEXP * PG1_S * 2);
  _Float16* pWpg2 = (_Float16*)alloc((size_t)HEXP * PG2_S * 2);
  _Float16* pAup  = (_Float16*)alloc((size_t)HEXP * AUP_S * 2);
  _Float16* pAdn  = (_Float16*)alloc((size_t)HEXP * ADN_S * 2);
  _Float16* pBup  = (_Float16*)alloc((size_t)HEXP * AUP_S * 2);
  _Float16* pWq   = (_Float16*)alloc((size_t)HH * WKV_S * 2);
  _Float16* pWk   = (_Float16*)alloc((size_t)HH * WKV_S * 2);
  _Float16* pWv   = (_Float16*)alloc((size_t)HH * WKV_S * 2);
  _Float16* pLAq  = (_Float16*)alloc((size_t)HEXP * LA_S * 2);
  _Float16* pLAk  = (_Float16*)alloc((size_t)HEXP * LA_S * 2);
  _Float16* pLAv  = (_Float16*)alloc((size_t)HEXP * LA_S * 2);
  _Float16* pWo   = (_Float16*)alloc((size_t)PRE_S * 2);

  // ---- weight packing (tiny; weights stay L2-resident afterwards) ----
  k_pack_phi<<<dim3(16, HEXP), 256, 0, stream>>>(ppre, ppost, pres, pPhi);
  k_pack<<<dim3(32, HEXP), 256, 0, stream>>>(Wpre, 256, 1, 256, 256, 8, 16,
                                             (long)256 * 256, pWpre);
  k_pack<<<dim3(32, HEXP), 256, 0, stream>>>(Wpg1, DPG, 1, 256, DPG, 8, 10,
                                             (long)256 * DPG, pWpg1);
  k_pack<<<dim3(32, HEXP), 256, 0, stream>>>(Wpg2, 256, 1, DPG, 256, 5, 16,
                                             (long)DPG * 256, pWpg2);
  k_pack<<<dim3(16, HEXP), 256, 0, stream>>>(aup, DALPHA, 1, 256, DALPHA, 8, 4,
                                             (long)256 * DALPHA, pAup);
  k_pack<<<dim3(4, HEXP), 256, 0, stream>>>(adn, DKP, 1, DALPHA, DKP, 2, 4,
                                            (long)DALPHA * DKP, pAdn);
  k_pack<<<dim3(16, HEXP), 256, 0, stream>>>(bup, DALPHA, 1, 256, DALPHA, 8, 4,
                                             (long)256 * DALPHA, pBup);
  k_pack<<<dim3(8, HH), 256, 0, stream>>>(Wq, 1, 256, 256, DK, 8, 2,
                                          (long)DK * 256, pWq);
  k_pack<<<dim3(8, HH), 256, 0, stream>>>(Wk, 1, 256, 256, DK, 8, 2,
                                          (long)DK * 256, pWk);
  k_pack<<<dim3(8, HH), 256, 0, stream>>>(Wv, 1, 256, 256, DV, 8, 2,
                                          (long)DV * 256, pWv);
  k_pack<<<dim3(4, HEXP), 256, 0, stream>>>(lAq, LR, 1, 256, LR, 8, 1,
                                            (long)256 * LR, pLAq);
  k_pack<<<dim3(4, HEXP), 256, 0, stream>>>(lAk, LR, 1, 256, LR, 8, 1,
                                            (long)256 * LR, pLAk);
  k_pack<<<dim3(4, HEXP), 256, 0, stream>>>(lAv, LR, 1, 256, LR, 8, 1,
                                            (long)256 * LR, pLAv);
  k_pack<<<dim3(32, 1), 256, 0, stream>>>(W_o, 1, 256, 256, 256, 8, 16,
                                          (long)0, pWo);

  k_zero<<<dim3(2048), dim3(256), 0, stream>>>(q_h, zcount);
  k_tab<<<dim3((TT * DK + 255) / 256), dim3(256), 0, stream>>>(freqs, pdel, cq, sq, ck, sk);
  k_prep<<<dim3(BT), dim3(256), 0, stream>>>(p_stream, qrw, kvrw, xs16, qsel, kvsel);
  k_expert<<<dim3(BT / 64, HEXP), dim3(256), 0, stream>>>(
      xs16, qsel, kvsel,
      pPhi, pWpre, pWpg1, pWpg2, pAup, pAdn, pBup, pWq, pWk, pWv, pLAq, pLAk, pLAv,
      lBq, lBk, lBv, bdn, mnorm, bpre, bpost, bres, apre, apost, ares, nw,
      cq, sq, ck, sk,
      q_h, k_h, v_h, a_h, b_h, acc_pre, acc_post, acc_Hres, acc_Hpost);
  k_scan<<<dim3(BB * HH), dim3(256), 0, stream>>>(q_h, k_h, v_h, a_h, b_h, out_cat);
  k_y<<<dim3(BT / 64), dim3(256), 0, stream>>>(acc_pre, out_cat, acc_post, pWo, yb);
  k_out<<<dim3(BT), dim3(256), 0, stream>>>(p_stream, acc_Hres, acc_Hpost, yb, out);
}
